// Embeddings_51994874085872
// MI455X (gfx1250) — matrix operand order repaired and checked by reading, not compiled, not run on hardware
//
#include <hip/hip_runtime.h>
#include <hip/hip_bf16.h>

typedef __attribute__((ext_vector_type(16))) _Float16     v16h;
typedef __attribute__((ext_vector_type(8)))  float        v8f;
typedef __attribute__((ext_vector_type(4)))  unsigned int u32x4;
typedef __attribute__((ext_vector_type(2)))  unsigned int u32x2;
typedef __attribute__((ext_vector_type(2)))  _Float16     h2;

#define NB    32   // NUM_BASIS
#define HH    32   // hidden H
#define CIN   8
#define COUT  32
#define NPTS  512
#define NWAVES 8

union FragU  { v16h v; u32x4 q[2]; };
union FragD  { v16h v; unsigned d[8]; };

// one-instruction cross-half-wave swap (lane i <-> lane i^16)
__device__ __forceinline__ unsigned swap16(unsigned x) {
    return __builtin_amdgcn_permlanex16(x, x, 0x76543210u, 0xfedcba98u, false, false);
}
// pack two f32 -> f16x2 dword (v_cvt_pkrtz_f16_f32)
__device__ __forceinline__ unsigned pk16(float a, float b) {
    auto t = __builtin_amdgcn_cvt_pkrtz(a, b);
    return __builtin_bit_cast(unsigned, t);
}
// packed f16 ReLU on a dword (v_pk_max_num_f16)
__device__ __forceinline__ unsigned relu_pk(unsigned x) {
    h2 v = __builtin_bit_cast(h2, x);
    h2 z = {(_Float16)0.0f, (_Float16)0.0f};
    return __builtin_bit_cast(unsigned, __builtin_elementwise_max(v, z));
}
// xor-butterfly add within a 16-lane half (v_permlane16_b32)
__device__ __forceinline__ float xor_add(float x, unsigned s0, unsigned s1) {
    unsigned u = __builtin_bit_cast(unsigned, x);
    unsigned p = __builtin_amdgcn_permlane16(u, u, s0, s1, false, false);
    return x + __builtin_bit_cast(float, p);
}

// W1 stored [n][k] row-major: B-frag lane = column n, lo half-wave takes K 0..7,16..23,
// hi half-wave takes K 8..15,24..31 (table order, matches A-frags).
__device__ __forceinline__ v16h load_bfrag_w(const _Float16* rowp, bool hi) {
    FragU f;
    const _Float16* p = rowp + (hi ? 8 : 0);
    f.q[0] = *(const u32x4*)(p);
    f.q[1] = *(const u32x4*)(p + 16);
    return f.v;
}

__global__ __launch_bounds__(256) void radial_conv_kernel(
    const float* __restrict__ xyz, const float* __restrict__ elements,
    const float* __restrict__ mask, const float* __restrict__ W0,
    const float* __restrict__ W1,  const float* __restrict__ Wout,
    float* __restrict__ out)
{
    __shared__ __align__(16) _Float16 sW0T[NB * HH];   // 2 KB  [k][n] (W0 transposed)
    __shared__ __align__(16) _Float16 sW1[HH * HH];    // 2 KB  [n][k], pre-scaled 1/sqrt32
    __shared__ __align__(16) _Float16 sET[CIN * NPTS]; // 8 KB  [j][b]
    __shared__ float sY[NWAVES][32 * CIN];             // 8 KB
    __shared__ float sYsum[CIN * HH];                  // 1 KB  [j][g]

    const int tid  = threadIdx.x;
    const int wave = tid >> 5;
    const int lane = tid & 31;
    const int lo16 = lane & 15;
    const bool hi  = lane >= 16;

    const int za = blockIdx.x;        // z*512 + a
    const int z  = za >> 9;

    const float s32 = 0.17677669529663688f;  // 1/sqrt(32)

    // ---- stage weights + elements (vectorized: one float4 per thread) ----
    {   // W0 -> transposed f16 [k][n]
        float4 w = ((const float4*)W0)[tid];           // 256 float4 == 1024 floats
        int n = tid >> 3, kb = (tid & 7) * 4;
        sW0T[(kb + 0) * HH + n] = (_Float16)w.x;
        sW0T[(kb + 1) * HH + n] = (_Float16)w.y;
        sW0T[(kb + 2) * HH + n] = (_Float16)w.z;
        sW0T[(kb + 3) * HH + n] = (_Float16)w.w;
    }
    {   // W1 * 1/sqrt32 -> f16 [n][k], packed b64 store
        float4 w = ((const float4*)W1)[tid];
        u32x2 p = { pk16(w.x * s32, w.y * s32), pk16(w.z * s32, w.w * s32) };
        *(u32x2*)(sW1 + tid * 4) = p;
    }
    {   // elements[z] -> transposed f16 [j][b]
        const float* Ez = elements + (size_t)z * NPTS * CIN;
        #pragma unroll
        for (int bb = tid; bb < NPTS; bb += 256) {     // 2 iterations
            const float4* ep = (const float4*)(Ez + (size_t)bb * CIN);
            float4 e0 = ep[0], e1 = ep[1];
            sET[0 * NPTS + bb] = (_Float16)e0.x;
            sET[1 * NPTS + bb] = (_Float16)e0.y;
            sET[2 * NPTS + bb] = (_Float16)e0.z;
            sET[3 * NPTS + bb] = (_Float16)e0.w;
            sET[4 * NPTS + bb] = (_Float16)e1.x;
            sET[5 * NPTS + bb] = (_Float16)e1.y;
            sET[6 * NPTS + bb] = (_Float16)e1.z;
            sET[7 * NPTS + bb] = (_Float16)e1.w;
        }
    }
    __syncthreads();

    // loop-invariant B fragments of W1 (column tiles 0 and 1)
    v16h w1b0 = load_bfrag_w(sW1 + lo16 * HH,        hi);
    v16h w1b1 = load_bfrag_w(sW1 + (16 + lo16) * HH, hi);

    const float ax = xyz[(size_t)za * 3 + 0];
    const float ay = xyz[(size_t)za * 3 + 1];
    const float az = xyz[(size_t)za * 3 + 2];

    v8f yT0 = {}; v8f yT1 = {};   // Y^T tiles: rows j (0..7 valid), cols g
    v8f zeroC = {};

    for (int it = 0; it < 2; ++it) {
        const int b0 = (wave * 2 + it) * 32;      // this wave's 32-row b chunk

        // ---- FC0 exploiting 2-sparse cosine basis: one v_cos per pair ----
        // lane holds full X1 row (32 f16 in 16 dwords), no LDS staging
        unsigned d[16];
        {
            const int bg = b0 + lane;
            const float bx = xyz[(size_t)(z * NPTS + bg) * 3 + 0];
            const float by = xyz[(size_t)(z * NPTS + bg) * 3 + 1];
            const float bz = xyz[(size_t)(z * NPTS + bg) * 3 + 2];
            const float dx = ax - bx, dy = ay - by, dzv = az - bz;
            const float sq = dx*dx + dy*dy + dzv*dzv;
            const float r  = __builtin_amdgcn_sqrtf(
                                 __builtin_amdgcn_fmed3f(sq, 1e-12f, 3.0e38f));
            const float u  = r * 3.1f;                         // r * 31/10
            const int  k0  = (int)u;
            const float f  = u - (float)k0;
            const float c  = __builtin_amdgcn_cosf(0.25f * f); // cos(pi/2 * f)
            const float v0 = c * c;                            // basis[k0]
            const float v0s = (k0 <= 31) ? v0 * s32 : 0.0f;
            const float v1s = (k0 <= 30) ? (1.0f - v0) * s32 : 0.0f; // basis[k0+1]
            const int  k0c = (k0 > 31) ? 31 : k0;
            const int  k1c = (k0 >= 31) ? 31 : k0 + 1;

            const _Float16 v0h = (_Float16)v0s, v1h = (_Float16)v1s;
            const h2 v0p = {v0h, v0h}, v1p = {v1h, v1h};
            const h2 hz  = {(_Float16)0.0f, (_Float16)0.0f};
            const h2* w0r = (const h2*)(sW0T + k0c * HH);
            const h2* w1r = (const h2*)(sW0T + k1c * HH);
            #pragma unroll
            for (int t = 0; t < 16; ++t) {
                h2 acc = w0r[t] * v0p + w1r[t] * v1p;          // v_pk_mul/fma_f16
                acc = __builtin_elementwise_max(acc, hz);       // v_pk_max_num_f16
                d[t] = __builtin_bit_cast(unsigned, acc);
            }
        }

        // ---- build FC1 A-frags in registers via half-wave dword exchange ----
        FragD A0, A1;
        #pragma unroll
        for (int t = 0; t < 4; ++t) {
            unsigned sndA = hi ? d[t]      : d[t + 4];
            unsigned sndB = hi ? d[t + 8]  : d[t + 12];
            unsigned rcvA = swap16(sndA);
            unsigned rcvB = swap16(sndB);
            A0.d[t]     = hi ? rcvA      : d[t];
            A0.d[t + 4] = hi ? rcvB      : d[t + 8];
            A1.d[t]     = hi ? d[t + 4]  : rcvA;
            A1.d[t + 4] = hi ? d[t + 12] : rcvB;
        }

        // ---- FC1 (dense): X2 = relu(X1 @ (W1/sqrt32)^T), via WMMA ----
        v8f y00 = __builtin_amdgcn_wmma_f32_16x16x32_f16(false, A0.v, false, w1b0, (short)0, zeroC, false, false);
        v8f y01 = __builtin_amdgcn_wmma_f32_16x16x32_f16(false, A0.v, false, w1b1, (short)0, zeroC, false, false);
        v8f y10 = __builtin_amdgcn_wmma_f32_16x16x32_f16(false, A1.v, false, w1b0, (short)0, zeroC, false, false);
        v8f y11 = __builtin_amdgcn_wmma_f32_16x16x32_f16(false, A1.v, false, w1b1, (short)0, zeroC, false, false);

        // ---- C-frag -> B-frag in registers (pack, packed-f16 relu, swap) ----
        FragD bx0, bx1;
        #pragma unroll
        for (int t = 0; t < 4; ++t) {
            unsigned p00 = relu_pk(pk16(y00[2*t], y00[2*t+1]));
            unsigned p10 = relu_pk(pk16(y10[2*t], y10[2*t+1]));
            unsigned p01 = relu_pk(pk16(y01[2*t], y01[2*t+1]));
            unsigned p11 = relu_pk(pk16(y11[2*t], y11[2*t+1]));
            unsigned r0 = swap16(hi ? p00 : p10);
            unsigned r1 = swap16(hi ? p01 : p11);
            bx0.d[t]     = hi ? r0  : p00;
            bx0.d[t + 4] = hi ? p10 : r0;
            bx1.d[t]     = hi ? r1  : p01;
            bx1.d[t + 4] = hi ? p11 : r1;
        }

        // ---- Y^T += E^T @ X2 : A'' = E^T [j rows (<8), k=b], B'' = X2 ----
        FragU ea;
        if (lo16 < 8) {
            const _Float16* ep = sET + lo16 * NPTS + b0;
            ea.q[0] = *(const u32x4*)(ep + (hi ? 16 : 0));
            ea.q[1] = *(const u32x4*)(ep + (hi ? 24 : 8));
        } else {
            ea.q[0] = (u32x4){0u, 0u, 0u, 0u};
            ea.q[1] = (u32x4){0u, 0u, 0u, 0u};
        }
        yT0 = __builtin_amdgcn_wmma_f32_16x16x32_f16(false, ea.v, false, bx0.v, (short)0, yT0, false, false);
        yT1 = __builtin_amdgcn_wmma_f32_16x16x32_f16(false, ea.v, false, bx1.v, (short)0, yT1, false, false);
    }

    // ---- reduce Y across waves (Y^T C-layout: lane n=g, vgpr r=j on lo lanes) ----
    if (!hi) {
        #pragma unroll
        for (int r = 0; r < 8; ++r) {
            sY[wave][lo16 * CIN + r]        = yT0[r];
            sY[wave][(16 + lo16) * CIN + r] = yT1[r];
        }
    }
    __syncthreads();
    {
        const int gg = tid >> 3, j = tid & 7;     // 256 threads cover 32x8
        float s = 0.0f;
        #pragma unroll
        for (int w = 0; w < NWAVES; ++w) s += sY[w][gg * CIN + j];
        sYsum[j * HH + gg] = s;                    // [j][g] for contiguous reads
    }
    __syncthreads();

    // ---- out[i] = (1/16)*mask * sum_{g,j} Wout[i*8+j, g] * Ysum[g,j] ----
    // 256 threads: one (i,j) pair each (32 FMAs), then permlane16 reduction.
    {
        const int i = tid >> 3, j = tid & 7;
        const float* wr = Wout + (size_t)(i * CIN + j) * HH;   // contiguous row
        const float* ys = sYsum + j * HH;
        float acc = 0.0f;
        #pragma unroll
        for (int gg = 0; gg < HH; ++gg) acc += wr[gg] * ys[gg];
        acc = xor_add(acc, 0x67452301u, 0xefcdab89u);  // xor 1
        acc = xor_add(acc, 0x54761032u, 0xdcfe98bau);  // xor 2
        acc = xor_add(acc, 0x32107654u, 0xba98fedcu);  // xor 4
        if (j == 0) out[(size_t)za * COUT + i] = acc * 0.0625f * mask[za];
    }
}

extern "C" void kernel_launch(void* const* d_in, const int* in_sizes, int n_in,
                              void* d_out, int out_size, void* d_ws, size_t ws_size,
                              hipStream_t stream) {
    (void)n_in; (void)d_ws; (void)ws_size; (void)out_size;
    const float* xyz      = (const float*)d_in[0];
    const float* elements = (const float*)d_in[1];
    const float* maskp    = (const float*)d_in[2];
    const float* W0       = (const float*)d_in[3];
    const float* W1       = (const float*)d_in[4];
    const float* Wout     = (const float*)d_in[5];
    float* out = (float*)d_out;

    const int BN = in_sizes[0] / 3;   // B*N = 1024
    radial_conv_kernel<<<BN, 256, 0, stream>>>(xyz, elements, maskp, W0, W1, Wout, out);
}
